// Attention_5153960755626
// MI455X (gfx1250) — compile-verified, matrix operand order checked by reading, not run on hardware
//
#include <hip/hip_runtime.h>
#include <hip/hip_bf16.h>

// CDNA5 (gfx1250) wave32 WMMA attention block.
typedef __bf16 bf16_t;
typedef __attribute__((ext_vector_type(16))) __bf16 v16bf;
typedef __attribute__((ext_vector_type(8)))  __bf16 v8bf;
typedef __attribute__((ext_vector_type(4)))  __bf16 v4bf;
typedef __attribute__((ext_vector_type(8)))  float  v8f;
typedef __attribute__((ext_vector_type(4)))  float  v4f;
typedef __attribute__((ext_vector_type(4)))  unsigned int v4u;
typedef __attribute__((ext_vector_type(8)))  int    v8i;
typedef __attribute__((ext_vector_type(4)))  int    v4i;

#define NTOK 4096
#define CDIM 768
#define NH   12
#define HD   64
#define QUIRK_SCALE 8.0f   // reference does q / (64**-0.5) == q * sqrt(64) == q * 8

#if __has_builtin(__builtin_amdgcn_tensor_load_to_lds) && \
    __has_builtin(__builtin_amdgcn_s_wait_tensorcnt)
#define USE_TDM 1
#else
#define USE_TDM 0
#endif

// Build a 16x32 (row x K) bf16 WMMA fragment from a row-major tile.
// ISA layout (CDNA5 ISA): lane&15 = row (A: M, B: N); per-lane
// element i maps to k = (i>>3)*16 + (lane>>4)*8 + (i&7) -> two contiguous
// 16-byte chunks at +kg*8 and +16+kg*8 (each a single ds_load_b128).
__device__ __forceinline__ v16bf frag16x32(const bf16_t* base, int stride) {
  const int lane = threadIdx.x & 31;
  const int r    = lane & 15;
  const int kg   = (lane >> 4) & 1;
  const bf16_t* p = base + r * stride + kg * 8;
  v8bf lo = *(const v8bf*)(p);
  v8bf hi = *(const v8bf*)(p + 16);
  v16bf f;
#pragma unroll
  for (int i = 0; i < 8; ++i) { f[i] = lo[i]; f[i + 8] = hi[i]; }
  return f;
}

__device__ __forceinline__ v8f wmma_bf16(v16bf a, v16bf b, v8f c) {
  // (neg_a, A, neg_b, B, c_mod, C, reuse_a, reuse_b)
  return __builtin_amdgcn_wmma_f32_16x16x32_bf16(false, a, false, b, (short)0, c,
                                                 false, false);
}

// ---------------------------------------------------------------------------
// Kernel 1: QKV projection. Y[N,3C] = X @ W^T + b written bf16 in [t][h][n][d].
// 8 waves, 128x64 tile per WG; each wave: 16 rows x 64 cols (A reused 4x).
// ---------------------------------------------------------------------------
__global__ __launch_bounds__(256) void qkv_gemm_kernel(
    const float* __restrict__ x, const float* __restrict__ qkv_w,
    const float* __restrict__ qkv_b, bf16_t* __restrict__ qkv_ws) {
  __shared__ bf16_t Xt[128 * 40];  // 128 rows x 32 K, stride 40
  __shared__ bf16_t Wt[64 * 40];
  const int tid   = threadIdx.x;
  const int w     = tid >> 5;            // 0..7 : 16-row strip
  const int mBase = blockIdx.x * 128;
  const int fBase = blockIdx.y * 64;

  v8f acc0 = {}, acc1 = {}, acc2 = {}, acc3 = {};
  for (int k0 = 0; k0 < CDIM; k0 += 32) {
    // Stage X (128x32) and W (64x32) as bf16: float4 load + packed cvt + b64 LDS store.
    for (int idx = tid; idx < 1024; idx += 256) {
      int r = idx >> 3, c4 = idx & 7;
      v4f xf = *(const v4f*)&x[(size_t)(mBase + r) * CDIM + k0 + c4 * 4];
      *(v4bf*)&Xt[r * 40 + c4 * 4] = __builtin_convertvector(xf, v4bf);
    }
    for (int idx = tid; idx < 512; idx += 256) {
      int r = idx >> 3, c4 = idx & 7;
      v4f wf = *(const v4f*)&qkv_w[(size_t)(fBase + r) * CDIM + k0 + c4 * 4];
      *(v4bf*)&Wt[r * 40 + c4 * 4] = __builtin_convertvector(wf, v4bf);
    }
    __syncthreads();
    v16bf a = frag16x32(&Xt[(w * 16) * 40], 40);
    acc0 = wmma_bf16(a, frag16x32(&Wt[0 * 16 * 40], 40), acc0);
    acc1 = wmma_bf16(a, frag16x32(&Wt[1 * 16 * 40], 40), acc1);
    acc2 = wmma_bf16(a, frag16x32(&Wt[2 * 16 * 40], 40), acc2);
    acc3 = wmma_bf16(a, frag16x32(&Wt[3 * 16 * 40], 40), acc3);
    __syncthreads();
  }

  // D-matrix layout: lane&15 = N, vgpr r -> M = (lane>>4)*8 + r.
  const int lane = tid & 31;
  const int n  = lane & 15;
  const int mo = (lane >> 4) * 8;
  v8f accs[4] = {acc0, acc1, acc2, acc3};
#pragma unroll
  for (int j = 0; j < 4; ++j) {
    int f   = fBase + j * 16 + n;   // feature in [0, 2304)
    int t   = f / CDIM;             // 0=q 1=k 2=v
    int rem = f - t * CDIM;
    int h = rem >> 6, d = rem & 63;
    float bias = qkv_b[f];
#pragma unroll
    for (int r = 0; r < 8; ++r) {
      int row = mBase + w * 16 + mo + r;
      qkv_ws[((size_t)(t * NH + h) * NTOK + row) * HD + d] =
          (bf16_t)(accs[j][r] + bias);
    }
  }
}

// ---------------------------------------------------------------------------
// Kernel 2: flash attention per head. 4 waves, 64 query rows per WG, streaming
// 32-key KV blocks through LDS with online softmax. K tile staged by the
// Tensor Data Mover when the builtin exists, else vectorized copies.
// ---------------------------------------------------------------------------
__global__ __launch_bounds__(128) void flash_attn_kernel(
    const bf16_t* __restrict__ qkv_ws, bf16_t* __restrict__ attn_ws) {
  __shared__ bf16_t Kt[32 * 64];      // key rows (row-major, packed, stride 64)
  __shared__ bf16_t Vt[64 * 40];      // V transposed: [d][key], stride 40
  __shared__ bf16_t Pt[4][16 * 40];   // per-wave P tile (16x32), stride 40
  const int tid  = threadIdx.x;
  const int w    = tid >> 5;
  const int lane = tid & 31;
  const int h     = blockIdx.y;
  const int qRow0 = blockIdx.x * 64 + w * 16;

  const bf16_t* Qg = qkv_ws + (size_t)(0 * NH + h) * NTOK * HD;
  const bf16_t* Kg = qkv_ws + (size_t)(1 * NH + h) * NTOK * HD;
  const bf16_t* Vg = qkv_ws + (size_t)(2 * NH + h) * NTOK * HD;

  // Q fragments (16 rows x 64 d = two K=32 steps) stay in registers.
  v16bf qa0 = frag16x32(Qg + (size_t)qRow0 * HD, HD);
  v16bf qa1 = frag16x32(Qg + (size_t)qRow0 * HD + 32, HD);

#if USE_TDM
  // LDS byte offset of Kt (AS(3) ptrtoint == LDS offset).
  const unsigned kt_lds = (unsigned)(size_t)(__attribute__((address_space(3))) bf16_t*)Kt;
#endif

  float mrun[8], lrun[8];
#pragma unroll
  for (int r = 0; r < 8; ++r) { mrun[r] = -1e30f; lrun[r] = 0.0f; }
  v8f o0 = {}, o1 = {}, o2 = {}, o3 = {};

  for (int kb = 0; kb < NTOK; kb += 32) {
#if USE_TDM
    if (w == 0) {
      // Tensor DMA: 2D tensor HD x NTOK (bf16), tile 64 x 32 rows -> Kt packed.
      // D# per CDNA5 ISA.
      unsigned long long ga = (unsigned long long)(size_t)(Kg + (size_t)kb * HD);
      v4u g0;
      g0.x = 1u;                                   // count=1, user descriptor
      g0.y = kt_lds;                               // lds_addr
      g0.z = (unsigned)(ga & 0xffffffffu);         // global_addr[31:0]
      g0.w = (unsigned)((ga >> 32) & 0x01ffffffu)  // global_addr[56:32]
             | 0x80000000u;                        // type=2 ("image")
      v8i g1;
      g1[0] = 0x00010000;                          // workgroup_mask=0, data_size=1 (2B)
      g1[1] = (HD & 0xffff) << 16;                 // tensor_dim0 = 64 (low 16)
      g1[2] = (NTOK & 0xffff) << 16;               // dim0 hi=0 | tensor_dim1 = 4096 lo
      g1[3] = (HD & 0xffff) << 16;                 // dim1 hi=0 | tile_dim0 = 64
      g1[4] = 32;                                  // tile_dim1 = 32, tile_dim2 = 0
      g1[5] = HD;                                  // tensor_dim0_stride = 64
      g1[6] = 0;                                   // stride0 hi | dim1_stride lo
      g1[7] = 0;
      v4i gz4 = {0, 0, 0, 0};
      v8i gz8 = {0, 0, 0, 0, 0, 0, 0, 0};
      // 6-arg variant (amdgpu-toolchain clang-23 / therock-10.0 headers).
      __builtin_amdgcn_tensor_load_to_lds(g0, g1, gz4, gz4, gz8, 0);
      __builtin_amdgcn_s_wait_tensorcnt(0);
    }
#else
    // Stage K tile row-major with 16-byte copies (2 per thread).
    for (int q = tid; q < 32 * 8; q += 128) {
      int r = q >> 3, c8 = q & 7;
      *(uint4*)&Kt[r * 64 + c8 * 8] =
          *(const uint4*)&Kg[(size_t)(kb + r) * HD + c8 * 8];
    }
#endif
    // Stage V transposed so PV B-fragments read contiguous keys.
    for (int idx = tid; idx < 32 * 64; idx += 128) {
      int k = idx >> 6, d = idx & 63;
      Vt[d * 40 + k] = Vg[(size_t)(kb + k) * HD + d];
    }
    __syncthreads();

    // S = Q K^T : B[dk][key] fragments come from K rows (contiguous d).
    v8f s0 = {}, s1 = {};
    s0 = wmma_bf16(qa0, frag16x32(&Kt[0], 64), s0);
    s0 = wmma_bf16(qa1, frag16x32(&Kt[32], 64), s0);
    s1 = wmma_bf16(qa0, frag16x32(&Kt[16 * 64], 64), s1);
    s1 = wmma_bf16(qa1, frag16x32(&Kt[16 * 64 + 32], 64), s1);

    // Online softmax: rows live as (lane>>4)*8 + r, cols across 16 lanes.
    float p0[8], p1[8], alpha[8];
#pragma unroll
    for (int r = 0; r < 8; ++r) {
      float a = s0[r] * QUIRK_SCALE;
      float b = s1[r] * QUIRK_SCALE;
      float mx = fmaxf(a, b);
#pragma unroll
      for (int msk = 1; msk <= 8; msk <<= 1)
        mx = fmaxf(mx, __shfl_xor(mx, msk, 32));
      float mnew = fmaxf(mrun[r], mx);
      alpha[r] = __expf(mrun[r] - mnew);
      mrun[r]  = mnew;
      p0[r] = __expf(a - mnew);
      p1[r] = __expf(b - mnew);
      float rs = p0[r] + p1[r];
#pragma unroll
      for (int msk = 1; msk <= 8; msk <<= 1) rs += __shfl_xor(rs, msk, 32);
      lrun[r] = lrun[r] * alpha[r] + rs;
    }
#pragma unroll
    for (int r = 0; r < 8; ++r) {
      o0[r] *= alpha[r]; o1[r] *= alpha[r];
      o2[r] *= alpha[r]; o3[r] *= alpha[r];
    }

    // D-layout -> A-layout conversion for P via this wave's LDS slab.
    {
      bf16_t* P = Pt[w];
      int nn = lane & 15, mo = (lane >> 4) * 8;
#pragma unroll
      for (int r = 0; r < 8; ++r) {
        P[(mo + r) * 40 + nn]      = (bf16_t)p0[r];
        P[(mo + r) * 40 + nn + 16] = (bf16_t)p1[r];
      }
    }
    v16bf pa = frag16x32(Pt[w], 40);
    // O += P V : B[key][d] fragments from transposed V (contiguous keys).
    o0 = wmma_bf16(pa, frag16x32(&Vt[0], 40), o0);
    o1 = wmma_bf16(pa, frag16x32(&Vt[16 * 40], 40), o1);
    o2 = wmma_bf16(pa, frag16x32(&Vt[32 * 40], 40), o2);
    o3 = wmma_bf16(pa, frag16x32(&Vt[48 * 40], 40), o3);
    __syncthreads();
  }

  // Normalize by row sums and store bf16 [N][C] for the projection GEMM.
  int nn = lane & 15, mo = (lane >> 4) * 8;
#pragma unroll
  for (int r = 0; r < 8; ++r) {
    float inv = 1.0f / lrun[r];
    int row = qRow0 + mo + r;
    bf16_t* dst = attn_ws + (size_t)row * CDIM + h * HD;
    dst[0 * 16 + nn] = (bf16_t)(o0[r] * inv);
    dst[1 * 16 + nn] = (bf16_t)(o1[r] * inv);
    dst[2 * 16 + nn] = (bf16_t)(o2[r] * inv);
    dst[3 * 16 + nn] = (bf16_t)(o3[r] * inv);
  }
}

// ---------------------------------------------------------------------------
// Kernel 3: output projection. out[N,C] = attn @ proj_w^T + proj_b (fp32 out).
// Same 128x64 tiling as kernel 1; A is already bf16 (straight b64 copies).
// ---------------------------------------------------------------------------
__global__ __launch_bounds__(256) void proj_gemm_kernel(
    const bf16_t* __restrict__ attn_ws, const float* __restrict__ proj_w,
    const float* __restrict__ proj_b, float* __restrict__ out) {
  __shared__ bf16_t At[128 * 40];
  __shared__ bf16_t Wt[64 * 40];
  const int tid   = threadIdx.x;
  const int w     = tid >> 5;
  const int mBase = blockIdx.x * 128;
  const int fBase = blockIdx.y * 64;

  v8f acc0 = {}, acc1 = {}, acc2 = {}, acc3 = {};
  for (int k0 = 0; k0 < CDIM; k0 += 32) {
    for (int idx = tid; idx < 1024; idx += 256) {
      int r = idx >> 3, c4 = idx & 7;
      *(uint2*)&At[r * 40 + c4 * 4] =
          *(const uint2*)&attn_ws[(size_t)(mBase + r) * CDIM + k0 + c4 * 4];
    }
    for (int idx = tid; idx < 512; idx += 256) {
      int r = idx >> 3, c4 = idx & 7;
      v4f wf = *(const v4f*)&proj_w[(size_t)(fBase + r) * CDIM + k0 + c4 * 4];
      *(v4bf*)&Wt[r * 40 + c4 * 4] = __builtin_convertvector(wf, v4bf);
    }
    __syncthreads();
    v16bf a = frag16x32(&At[(w * 16) * 40], 40);
    acc0 = wmma_bf16(a, frag16x32(&Wt[0 * 16 * 40], 40), acc0);
    acc1 = wmma_bf16(a, frag16x32(&Wt[1 * 16 * 40], 40), acc1);
    acc2 = wmma_bf16(a, frag16x32(&Wt[2 * 16 * 40], 40), acc2);
    acc3 = wmma_bf16(a, frag16x32(&Wt[3 * 16 * 40], 40), acc3);
    __syncthreads();
  }

  const int lane = tid & 31;
  const int n  = lane & 15;
  const int mo = (lane >> 4) * 8;
  v8f accs[4] = {acc0, acc1, acc2, acc3};
#pragma unroll
  for (int j = 0; j < 4; ++j) {
    int f = fBase + j * 16 + n;
    float bias = proj_b[f];
#pragma unroll
    for (int r = 0; r < 8; ++r) {
      int row = mBase + w * 16 + mo + r;
      out[(size_t)row * CDIM + f] = accs[j][r] + bias;
    }
  }
}

// ---------------------------------------------------------------------------
extern "C" void kernel_launch(void* const* d_in, const int* in_sizes, int n_in,
                              void* d_out, int out_size, void* d_ws, size_t ws_size,
                              hipStream_t stream) {
  const float* x      = (const float*)d_in[0];
  const float* qkv_w  = (const float*)d_in[1];
  const float* qkv_b  = (const float*)d_in[2];
  const float* proj_w = (const float*)d_in[3];
  const float* proj_b = (const float*)d_in[4];
  float* out = (float*)d_out;

  // Workspace: bf16 QKV in [t][h][n][d] (18.9 MB) + bf16 attention out (6.3 MB).
  bf16_t* qkv_ws  = (bf16_t*)d_ws;
  bf16_t* attn_ws = qkv_ws + (size_t)3 * NH * NTOK * HD;

  qkv_gemm_kernel<<<dim3(NTOK / 128, (3 * CDIM) / 64), 256, 0, stream>>>(
      x, qkv_w, qkv_b, qkv_ws);
  flash_attn_kernel<<<dim3(NTOK / 64, NH), 128, 0, stream>>>(qkv_ws, attn_ws);
  proj_gemm_kernel<<<dim3(NTOK / 128, CDIM / 64), 256, 0, stream>>>(
      attn_ws, proj_w, proj_b, out);
}
